// HierarchicalGAT_81604378624409
// MI455X (gfx1250) — compile-verified, matrix operand order checked by reading, not run on hardware
//
#include <hip/hip_runtime.h>

// ---------------------------------------------------------------------------
// HierarchicalGAT fused forward for MI455X (gfx1250).
// ~94 GFLOP vs ~290MB HBM traffic -> memory bound only on the bf16 WMMA path
// (v_wmma_f32_16x16x32_bf16). One block = 16 batch rows (WMMA M-tile), 8
// waves, each wave owns one 16-wide N-tile of every GEMM. Weights converted
// once fp32->bf16 into d_ws (L2-resident ~1MB). Weight fragments are hoisted
// into registers per stage and reused across node iterations (3-5x less L2
// weight traffic than re-streaming per GEMM); activation fragments are loaded
// once and feed both K and V WMMAs. Intermediates live in a ~106KB LDS arena.
// ---------------------------------------------------------------------------

typedef __attribute__((ext_vector_type(16))) __bf16 v16bf;
typedef __attribute__((ext_vector_type(8)))  float  v8f;
typedef unsigned short u16;

// bf16 weight copies inside d_ws (offsets in u16 elements)
enum : size_t {
  WS_G_IN  = 0,        // g_in_w      384x128
  WS_S_IN  = 49152,    // s_self_in_w 384x128
  WS_C_IN  = 98304,    // s_cross_in_w
  WS_D_IN  = 147456,   // d_in_w
  WS_G_OUT = 196608,   // 128x128 each below
  WS_S_OUT = 212992,
  WS_C_OUT = 229376,
  WS_D_OUT = 245760,
  WS_CTX   = 262144,
  WS_WL    = 278528,
  WS_WR    = 294912,
  WS_ACT   = 311296,
  WS_F1    = 327680,   // f_w1 256x512
  WS_F2    = 458752,   // f_w2 128x256
  WS_TOTAL = 491520
};

// GAT edges
__constant__ int cSRC[10] = {0,1,2,3,0,2,1,4,3,4};
// incoming-edge pairs per target (TGT = 1,0,3,2,2,0,4,1,4,3)
__constant__ int cE0[5] = {1,0,3,2,6};
__constant__ int cE1[5] = {5,7,4,9,8};

struct KArgs {
  const float* x[17];
  const float* p[43];
  const u16*   ws;
  float*       out;
};

__device__ __forceinline__ u16 f2bf(float f) {
  unsigned u = __float_as_uint(f);
  u += 0x7fffu + ((u >> 16) & 1u);
  return (u16)(u >> 16);
}
__device__ __forceinline__ float bf2f(u16 h) {
  return __uint_as_float(((unsigned)h) << 16);
}

// WMMA 16x32 bf16 fragment per ISA 7.12.2: lane&15 = row, lane>>4 selects
// K 0-7/16-23 vs 8-15/24-31. Row-major source, row stride ld (elements).
__device__ __forceinline__ v16bf ldfrag(const u16* m, int ld, int r0, int k0) {
  int lane = threadIdx.x & 31;
  const u16* p = m + (size_t)(r0 + (lane & 15)) * ld + k0 + ((lane >> 4) << 3);
  union { v16bf v; uint4 q[2]; } f;
  f.q[0] = *(const uint4*)(p);
  f.q[1] = *(const uint4*)(p + 16);
  return f.v;
}

// Full-K operand held in registers: K/32 fragments (K=128 -> 32 VGPRs).
template <int K> struct Frag { v16bf f[K / 32]; };

template <int K>
__device__ __forceinline__ Frag<K> ldmat(const u16* M, int ld, int r0) {
  Frag<K> o;
#pragma unroll
  for (int i = 0; i < K / 32; ++i) o.f[i] = ldfrag(M, ld, r0, i * 32);
  return o;
}

template <int K>
__device__ __forceinline__ v8f wmma_acc(const Frag<K>& A, const Frag<K>& B) {
  v8f acc = {0.f, 0.f, 0.f, 0.f, 0.f, 0.f, 0.f, 0.f};
#pragma unroll
  for (int i = 0; i < K / 32; ++i)
    acc = __builtin_amdgcn_wmma_f32_16x16x32_bf16(false, A.f[i], false, B.f[i],
                                                  (short)0, acc, false, false);
  return acc;
}

// one-shot Y(16x16) = A(16xK) @ W(NxK)^T for this wave's N-tile (col n0)
template <int K>
__device__ __forceinline__ v8f gemmK(const u16* A, int ldA, const u16* W, int ldW, int n0) {
  Frag<K> a = ldmat<K>(A, ldA, 0);
  Frag<K> b = ldmat<K>(W, ldW, n0);
  return wmma_acc<K>(a, b);
}

// C/D layout: vgpr j, lane l: row = j + (l>=16)*8, col = l&15.
__device__ __forceinline__ void st_f32(float* Y, int ldY, int n0, v8f acc, const float* bias) {
  int lane = threadIdx.x & 31;
  int col = n0 + (lane & 15);
  int r0 = (lane >> 4) << 3;
  float b = bias ? bias[col] : 0.f;
#pragma unroll
  for (int j = 0; j < 8; ++j) Y[(r0 + j) * ldY + col] = acc[j] + b;
}
__device__ __forceinline__ void st_bf16(u16* Y, int ldY, int n0, v8f acc, const float* bias, int relu) {
  int lane = threadIdx.x & 31;
  int col = n0 + (lane & 15);
  int r0 = (lane >> 4) << 3;
  float b = bias ? bias[col] : 0.f;
#pragma unroll
  for (int j = 0; j < 8; ++j) {
    float v = acc[j] + b;
    if (relu) v = v > 0.f ? v : 0.f;
    Y[(r0 + j) * ldY + col] = f2bf(v);
  }
}

// row-wise LayerNorm of f32 tile [16][128] (+ optional bf16 residual),
// bf16 result to out[:, col0:col0+128] with stride ldo. 256 threads.
__device__ __forceinline__ void lnorm(const float* X, const u16* res, const float* g,
                                      const float* bt, u16* out, int ldo, int col0,
                                      float* red, float* stat) {
  int row = threadIdx.x >> 4, p = threadIdx.x & 15;
  float s = 0.f;
  for (int c = p; c < 128; c += 16)
    s += X[row * 128 + c] + (res ? bf2f(res[row * 128 + c]) : 0.f);
  red[row * 16 + p] = s;
  __syncthreads();
  if (p == 0) {
    float t = 0.f;
    for (int i = 0; i < 16; ++i) t += red[row * 16 + i];
    stat[row] = t * (1.f / 128.f);
  }
  __syncthreads();
  float mu = stat[row];
  s = 0.f;
  for (int c = p; c < 128; c += 16) {
    float v = X[row * 128 + c] + (res ? bf2f(res[row * 128 + c]) : 0.f) - mu;
    s += v * v;
  }
  red[row * 16 + p] = s;
  __syncthreads();
  if (p == 0) {
    float t = 0.f;
    for (int i = 0; i < 16; ++i) t += red[row * 16 + i];
    stat[16 + row] = rsqrtf(t * (1.f / 128.f) + 1e-5f);
  }
  __syncthreads();
  float inv = stat[16 + row];
  for (int c = p; c < 128; c += 16) {
    float v = X[row * 128 + c] + (res ? bf2f(res[row * 128 + c]) : 0.f);
    out[row * ldo + col0 + c] = f2bf((v - mu) * inv * g[c] + bt[c]);
  }
  __syncthreads();
}

__device__ __forceinline__ void load_node(const float* x, int b0, u16* dst) {
  for (int i = threadIdx.x; i < 2048; i += 256)
    dst[i] = f2bf(x[(size_t)(b0 + (i >> 7)) * 128 + (i & 127)]);
}
__device__ __forceinline__ void load_node_add(const float* x, int b0, const float* addF, u16* dst) {
  for (int i = threadIdx.x; i < 2048; i += 256)
    dst[i] = f2bf(x[(size_t)(b0 + (i >> 7)) * 128 + (i & 127)] + addF[i]);
}
__device__ __forceinline__ void tile_to_bf16(const float* X, u16* out, int ldo, int col0) {
  for (int i = threadIdx.x; i < 2048; i += 256)
    out[(i >> 7) * ldo + col0 + (i & 127)] = f2bf(X[i]);
}

// out[128] = b + qv(1x128) @ W(128x128)^T  (constant query projection)
__device__ __forceinline__ void const_qproj(const float* qv, const float* W, const float* b, float* out) {
  int t = threadIdx.x;
  if (t < 128) {
    float s = b[t];
    for (int k = 0; k < 128; ++k) s += qv[k] * W[t * 128 + k];
    out[t] = s;
  }
}

// scores[(m*4+h)*8+j] = scale * q[m,h,:] . K_j[m,h,:]   (q is bf16 tile)
__device__ __forceinline__ void scores_tile(const u16* Q, const u16* Kb, int J, float* esc) {
  int total = 16 * 4 * J;
  for (int idx = threadIdx.x; idx < total; idx += 256) {
    int m = idx / (4 * J), r = idx % (4 * J), h = r / J, j = r % J;
    const u16* q = Q + m * 128 + h * 32;
    const u16* k = Kb + j * 2048 + m * 128 + h * 32;
    float s = 0.f;
#pragma unroll
    for (int d = 0; d < 32; ++d) s += bf2f(q[d]) * bf2f(k[d]);
    esc[(m * 4 + h) * 8 + j] = s * 0.17677669529663687f; // 1/sqrt(32)
  }
}
// same with a broadcast f32 query vector (constant learned query)
__device__ __forceinline__ void scores_const(const float* q, const u16* Kb, int J, float* esc) {
  int total = 16 * 4 * J;
  for (int idx = threadIdx.x; idx < total; idx += 256) {
    int m = idx / (4 * J), r = idx % (4 * J), h = r / J, j = r % J;
    const u16* k = Kb + j * 2048 + m * 128 + h * 32;
    float s = 0.f;
#pragma unroll
    for (int d = 0; d < 32; ++d) s += q[h * 32 + d] * bf2f(k[d]);
    esc[(m * 4 + h) * 8 + j] = s * 0.17677669529663687f;
  }
}
__device__ __forceinline__ void softmax_scores(float* esc, int J) {
  int t = threadIdx.x;
  if (t < 64) {
    float* s = esc + t * 8;
    float mx = s[0];
    for (int j = 1; j < J; ++j) mx = fmaxf(mx, s[j]);
    float den = 0.f;
    for (int j = 0; j < J; ++j) { float e = __expf(s[j] - mx); s[j] = e; den += e; }
    float r = 1.f / den;
    for (int j = 0; j < J; ++j) s[j] *= r;
  }
}
// O[m,n] (bf16) = sum_j alpha[m,h(n),j] * V_j[m,n]
__device__ __forceinline__ void attn_out(const float* esc, const u16* Vb, int J, u16* O) {
  for (int i = threadIdx.x; i < 2048; i += 256) {
    int m = i >> 7, n = i & 127, h = n >> 5;
    const float* al = esc + (m * 4 + h) * 8;
    float s = 0.f;
    for (int j = 0; j < J; ++j) s += al[j] * bf2f(Vb[j * 2048 + m * 128 + n]);
    O[i] = f2bf(s);
  }
}

// ---------------------------------------------------------------------------
// LDS arena layout (bytes)
// ---------------------------------------------------------------------------
enum : unsigned {
  OFF_HCAT = 0,        // u16 [16][512]  h_global|h_state|h_actor|h_dyn
  OFF_BUFA = 16384,    // u16 [5][16][128] stage activations (in-place updated)
  OFF_BUFK = 36864,    // u16 [5][16][128] K-proj / xl / FFN hidden
  OFF_BUFV = 57344,    // u16 [5][16][128] V-proj / xr
  OFF_TILF = 77824,    // f32 [16][128]  GEMM f32 output / LN input
  OFF_TILB = 86016,    // u16 [16][128]
  OFF_TILB2 = 90112,   // u16 [16][128]
  OFF_CTXF = 94208,    // f32 [16][128]  ctx / o_mean accumulator
  OFF_VEC  = 102400,   // f32 [512]  qp_g | qp2 | a_att | lw
  OFF_ESC  = 104448,   // f32 [640]  attention / edge scores
  OFF_RED  = 107008,   // f32 [256]
  OFF_STAT = 108032,   // f32 [64]
  SMEM_BYTES = 108288
};

__global__ void __launch_bounds__(256) hgat_fused(KArgs A) {
  extern __shared__ char smem[];
  u16*   hcat   = (u16*)(smem + OFF_HCAT);
  u16*   bufA   = (u16*)(smem + OFF_BUFA);
  u16*   bufK   = (u16*)(smem + OFF_BUFK);
  u16*   bufV   = (u16*)(smem + OFF_BUFV);
  float* tileF  = (float*)(smem + OFF_TILF);
  u16*   tileB  = (u16*)(smem + OFF_TILB);
  u16*   tileB2 = (u16*)(smem + OFF_TILB2);
  float* ctxF   = (float*)(smem + OFF_CTXF);
  float* vecs   = (float*)(smem + OFF_VEC);
  float* esc    = (float*)(smem + OFF_ESC);
  float* red    = (float*)(smem + OFF_RED);
  float* stat   = (float*)(smem + OFF_STAT);

  const int wave = threadIdx.x >> 5;
  const int n0   = wave * 16;
  const int b0   = blockIdx.x * 16;
  const u16* ws  = A.ws;

  // ---- block-constant precompute: qp_g, qp2, a_att copy, layer weights ----
  const_qproj(A.p[0],  A.p[1], A.p[2], vecs);        // g_query @ Wq_g
  const_qproj(A.p[19], A.p[7], A.p[8], vecs + 128);  // s_query @ Wq_s
  if (threadIdx.x < 128) vecs[256 + threadIdx.x] = A.p[24][threadIdx.x]; // a_att
  if (threadIdx.x == 0) {
    const float* li = A.p[42];
    float mx = fmaxf(fmaxf(li[0], li[1]), fmaxf(li[2], li[3]));
    float e0 = __expf(li[0] - mx), e1 = __expf(li[1] - mx);
    float e2 = __expf(li[2] - mx), e3 = __expf(li[3] - mx);
    float d = e0 + e1 + e2 + e3;
    vecs[384] = e0 / d; vecs[385] = e1 / d; vecs[386] = e2 / d; vecs[387] = e3 / d;
  }

  // ======================= Stage G: global MHA ===========================
  load_node(A.x[0], b0, bufA);
  load_node(A.x[1], b0, bufA + 2048);
  load_node(A.x[2], b0, bufA + 4096);
  __syncthreads();
  { // hoisted weight fragments, reused across 3 nodes; A loaded once per node
    Frag<128> Bk = ldmat<128>(ws + WS_G_IN + 128 * 128, 128, n0);
    Frag<128> Bv = ldmat<128>(ws + WS_G_IN + 256 * 128, 128, n0);
    for (int j = 0; j < 3; ++j) {
      Frag<128> Aj = ldmat<128>(bufA + j * 2048, 128, 0);
      st_bf16(bufK + j * 2048, 128, n0, wmma_acc<128>(Aj, Bk), A.p[2] + 128, 0);
      st_bf16(bufV + j * 2048, 128, n0, wmma_acc<128>(Aj, Bv), A.p[2] + 256, 0);
    }
  }
  __syncthreads();
  scores_const(vecs, bufK, 3, esc);
  __syncthreads();
  softmax_scores(esc, 3);
  __syncthreads();
  attn_out(esc, bufV, 3, tileB);
  __syncthreads();
  { v8f o = gemmK<128>(tileB, 128, ws + WS_G_OUT, 128, n0);
    st_f32(tileF, 128, n0, o, A.p[4]); }
  __syncthreads();
  lnorm(tileF, nullptr, A.p[5], A.p[6], hcat, 512, 0, red, stat); // h_global

  // ======================= Stage S: self MHA over 5 ======================
  for (int i = 0; i < 5; ++i) load_node(A.x[3 + i], b0, bufA + i * 2048);
  __syncthreads();
  {
    Frag<128> Bk = ldmat<128>(ws + WS_S_IN + 128 * 128, 128, n0);
    Frag<128> Bv = ldmat<128>(ws + WS_S_IN + 256 * 128, 128, n0);
    for (int i = 0; i < 5; ++i) {
      Frag<128> Ai = ldmat<128>(bufA + i * 2048, 128, 0);
      st_bf16(bufK + i * 2048, 128, n0, wmma_acc<128>(Ai, Bk), A.p[8] + 128, 0);
      st_bf16(bufV + i * 2048, 128, n0, wmma_acc<128>(Ai, Bv), A.p[8] + 256, 0);
    }
  }
  __syncthreads();
  {
    Frag<128> Bq = ldmat<128>(ws + WS_S_IN, 128, n0);
    Frag<128> Bo = ldmat<128>(ws + WS_S_OUT, 128, n0);
    for (int i = 0; i < 5; ++i) {
      Frag<128> Ai = ldmat<128>(bufA + i * 2048, 128, 0);
      st_bf16(tileB, 128, n0, wmma_acc<128>(Ai, Bq), A.p[8], 0);
      __syncthreads();
      scores_tile(tileB, bufK, 5, esc);
      __syncthreads();
      softmax_scores(esc, 5);
      __syncthreads();
      attn_out(esc, bufV, 5, tileB2);
      __syncthreads();
      Frag<128> Ao = ldmat<128>(tileB2, 128, 0);
      st_f32(tileF, 128, n0, wmma_acc<128>(Ao, Bo), A.p[10]);
      __syncthreads();
      // h_self_i = LN1(attn + s_nodes_i), overwrite in place
      lnorm(tileF, bufA + i * 2048, A.p[15], A.p[16], bufA + i * 2048, 128, 0, red, stat);
    }
  }

  // ============ Cross MHA: Sk=1 => softmax==1 => o == V(h_global) ========
  { v8f v = gemmK<128>(hcat, 512, ws + WS_C_IN + 256 * 128, 128, n0);
    st_bf16(tileB, 128, n0, v, A.p[12] + 256, 0); }
  __syncthreads();
  { v8f o = gemmK<128>(tileB, 128, ws + WS_C_OUT, 128, n0);
    st_f32(tileF, 128, n0, o, A.p[14]); }
  __syncthreads();
  for (int i = 0; i < 5; ++i)  // h_cross_i = LN2(c_out + h_self_i) in place
    lnorm(tileF, bufA + i * 2048, A.p[17], A.p[18], bufA + i * 2048, 128, 0, red, stat);

  // ===================== h_state: const-query MHA ========================
  {
    Frag<128> Bk = ldmat<128>(ws + WS_S_IN + 128 * 128, 128, n0);
    Frag<128> Bv = ldmat<128>(ws + WS_S_IN + 256 * 128, 128, n0);
    for (int i = 0; i < 5; ++i) {
      Frag<128> Ai = ldmat<128>(bufA + i * 2048, 128, 0);
      st_bf16(bufK + i * 2048, 128, n0, wmma_acc<128>(Ai, Bk), A.p[8] + 128, 0);
      st_bf16(bufV + i * 2048, 128, n0, wmma_acc<128>(Ai, Bv), A.p[8] + 256, 0);
    }
  }
  __syncthreads();
  scores_const(vecs + 128, bufK, 5, esc);
  __syncthreads();
  softmax_scores(esc, 5);
  __syncthreads();
  attn_out(esc, bufV, 5, tileB);
  __syncthreads();
  { v8f o = gemmK<128>(tileB, 128, ws + WS_S_OUT, 128, n0);
    st_f32(tileF, 128, n0, o, A.p[10]); }
  __syncthreads();
  tile_to_bf16(tileF, hcat, 512, 128);  // h_state (no LN)
  __syncthreads();
  { v8f c = gemmK<128>(hcat + 128, 512, ws + WS_CTX, 128, n0); // ctx
    st_f32(ctxF, 128, n0, c, A.p[27]); }
  __syncthreads();

  // ============================ GAT stage ================================
  for (int i = 0; i < 5; ++i) load_node_add(A.x[8 + i], b0, ctxF, bufA + i * 2048);
  __syncthreads();
  {
    Frag<128> Bl = ldmat<128>(ws + WS_WL, 128, n0);
    Frag<128> Br = ldmat<128>(ws + WS_WR, 128, n0);
    for (int i = 0; i < 5; ++i) {
      Frag<128> Ai = ldmat<128>(bufA + i * 2048, 128, 0);
      st_bf16(bufK + i * 2048, 128, n0, wmma_acc<128>(Ai, Bl), A.p[21], 0); // xl
      st_bf16(bufV + i * 2048, 128, n0, wmma_acc<128>(Ai, Br), A.p[23], 0); // xr
    }
  }
  __syncthreads();
  // e[e,m,h] = leaky_relu(xl[SRC]+xr[TGT]) . a_att[h]
  for (int idx = threadIdx.x; idx < 640; idx += 256) {
    int e = idx / 64, r = idx % 64, m = r >> 2, h = r & 3;
    const u16* xl = bufK + cSRC[e] * 2048 + m * 128 + h * 32;
    const int tgt[10] = {1, 0, 3, 2, 2, 0, 4, 1, 4, 3};
    const u16* xr = bufV + tgt[e] * 2048 + m * 128 + h * 32;
    float s = 0.f;
    for (int d = 0; d < 32; ++d) {
      float v = bf2f(xl[d]) + bf2f(xr[d]);
      v = v > 0.f ? v : 0.2f * v;
      s += v * vecs[256 + h * 32 + d];
    }
    esc[e * 64 + r] = s;
  }
  __syncthreads();
  // segment softmax over the 2 incoming edges of each target
  if (threadIdx.x < 64) {
    int m = threadIdx.x >> 2, h = threadIdx.x & 3;
    for (int tg = 0; tg < 5; ++tg) {
      int a = cE0[tg] * 64 + m * 4 + h, b = cE1[tg] * 64 + m * 4 + h;
      float ea = esc[a], eb = esc[b];
      float mx = fmaxf(ea, eb);
      float xa = __expf(ea - mx), xb = __expf(eb - mx);
      float d = 1.f / (xa + xb);
      esc[a] = xa * d; esc[b] = xb * d;
    }
  }
  __syncthreads();
  // mean over targets == sum over all 10 edges / 5  (+ a_bias)
  for (int i = threadIdx.x; i < 2048; i += 256) {
    int m = i >> 7, n = i & 127, h = n >> 5;
    float s = 0.f;
#pragma unroll
    for (int e = 0; e < 10; ++e)
      s += esc[e * 64 + m * 4 + h] * bf2f(bufK[cSRC[e] * 2048 + m * 128 + n]);
    tileF[i] = s * 0.2f + A.p[25][n];
  }
  __syncthreads();
  lnorm(tileF, nullptr, A.p[28], A.p[29], hcat, 512, 256, red, stat); // h_actor

  // ========================== Dynamics stage =============================
  { v8f c = gemmK<128>(hcat + 256, 512, ws + WS_ACT, 128, n0); // actor proj
    st_f32(tileF, 128, n0, c, A.p[35]); }
  __syncthreads();
  for (int i = 0; i < 4; ++i) load_node_add(A.x[13 + i], b0, tileF, bufA + i * 2048);
  __syncthreads();
  {
    Frag<128> Bk = ldmat<128>(ws + WS_D_IN + 128 * 128, 128, n0);
    Frag<128> Bv = ldmat<128>(ws + WS_D_IN + 256 * 128, 128, n0);
    for (int i = 0; i < 4; ++i) {
      Frag<128> Ai = ldmat<128>(bufA + i * 2048, 128, 0);
      st_bf16(bufK + i * 2048, 128, n0, wmma_acc<128>(Ai, Bk), A.p[31] + 128, 0);
      st_bf16(bufV + i * 2048, 128, n0, wmma_acc<128>(Ai, Bv), A.p[31] + 256, 0);
    }
  }
  for (int i = threadIdx.x; i < 2048; i += 256) ctxF[i] = 0.f; // o_mean acc
  __syncthreads();
  {
    Frag<128> Bq = ldmat<128>(ws + WS_D_IN, 128, n0);
    for (int i = 0; i < 4; ++i) {
      Frag<128> Ai = ldmat<128>(bufA + i * 2048, 128, 0);
      st_bf16(tileB, 128, n0, wmma_acc<128>(Ai, Bq), A.p[31], 0);
      __syncthreads();
      scores_tile(tileB, bufK, 4, esc);
      __syncthreads();
      softmax_scores(esc, 4);
      __syncthreads();
      for (int ii = threadIdx.x; ii < 2048; ii += 256) {
        int m = ii >> 7, n = ii & 127, h = n >> 5;
        const float* al = esc + (m * 4 + h) * 8;
        float s = 0.f;
        for (int j = 0; j < 4; ++j) s += al[j] * bf2f(bufV[j * 2048 + m * 128 + n]);
        ctxF[ii] += 0.25f * s; // mean over 4 nodes folded before out-proj
      }
      __syncthreads();
    }
  }
  tile_to_bf16(ctxF, tileB, 128, 0);
  __syncthreads();
  { v8f o = gemmK<128>(tileB, 128, ws + WS_D_OUT, 128, n0);
    st_f32(tileF, 128, n0, o, A.p[33]); }
  __syncthreads();
  lnorm(tileF, nullptr, A.p[36], A.p[37], hcat, 512, 384, red, stat); // h_dyn

  // =================== Final FFN + weighted skip =========================
  u16* hid = bufK; // [16][256] aliases bufK
  for (int rep = 0; rep < 2; ++rep) {
    int nn = (wave + rep * 8) * 16;
    v8f h = gemmK<512>(hcat, 512, ws + WS_F1, 512, nn);
    st_bf16(hid, 256, nn, h, A.p[39], 1); // relu
  }
  __syncthreads();
  {
    v8f o = gemmK<256>(hid, 256, ws + WS_F2, 256, n0);
    int lane = threadIdx.x & 31;
    int col = n0 + (lane & 15);
    int r0 = (lane >> 4) << 3;
    float bb = A.p[41][col];
    float lw0 = vecs[384], lw1 = vecs[385], lw2 = vecs[386], lw3 = vecs[387];
#pragma unroll
    for (int j = 0; j < 8; ++j) {
      int row = r0 + j;
      float hw = lw0 * bf2f(hcat[row * 512 + col]) +
                 lw1 * bf2f(hcat[row * 512 + 128 + col]) +
                 lw2 * bf2f(hcat[row * 512 + 256 + col]) +
                 lw3 * bf2f(hcat[row * 512 + 384 + col]);
      A.out[(size_t)(b0 + row) * 128 + col] = o[j] + bb + hw;
    }
  }
}

// fp32 -> bf16 weight conversion (once per launch; weights stay L2-hot)
__global__ void __launch_bounds__(256) conv_bf16(const float* __restrict__ s,
                                                 u16* __restrict__ d, int n) {
  int i = blockIdx.x * 256 + threadIdx.x;
  if (i < n) d[i] = f2bf(s[i]);
}

extern "C" void kernel_launch(void* const* d_in, const int* in_sizes, int n_in,
                              void* d_out, int out_size, void* d_ws, size_t ws_size,
                              hipStream_t stream) {
  (void)n_in; (void)out_size; (void)ws_size;
  u16* ws = (u16*)d_ws;

  struct Cv { int p; size_t off; int n; };
  const Cv cvt[] = {
    {1, WS_G_IN, 49152}, {7, WS_S_IN, 49152}, {11, WS_C_IN, 49152}, {30, WS_D_IN, 49152},
    {3, WS_G_OUT, 16384}, {9, WS_S_OUT, 16384}, {13, WS_C_OUT, 16384}, {32, WS_D_OUT, 16384},
    {26, WS_CTX, 16384}, {20, WS_WL, 16384}, {22, WS_WR, 16384}, {34, WS_ACT, 16384},
    {38, WS_F1, 131072}, {40, WS_F2, 32768}};
  for (const Cv& c : cvt)
    conv_bf16<<<(c.n + 255) / 256, 256, 0, stream>>>(
        (const float*)d_in[17 + c.p], ws + c.off, c.n);

  KArgs A;
  for (int i = 0; i < 17; ++i) A.x[i] = (const float*)d_in[i];
  for (int j = 0; j < 43; ++j) A.p[j] = (const float*)d_in[17 + j];
  A.ws = ws;
  A.out = (float*)d_out;

  int B = in_sizes[0] / 128;      // 32768
  int blocks = B / 16;            // 2048 blocks of 16 rows
  hgat_fused<<<blocks, 256, SMEM_BYTES, stream>>>(A);
}
